// Patient2Vec0_26353919328294
// MI455X (gfx1250) — matrix operand order repaired and checked by reading, not compiled, not run on hardware
//
#include <hip/hip_runtime.h>
#include <hip/hip_bf16.h>

// Model dims
#define B_ 32
#define S_ 128
#define P_ 32
#define I_ 512
#define E_ 512
#define H_ 512
#define O_ 10

typedef __bf16 v16bf __attribute__((ext_vector_type(16)));
typedef __bf16 v8bf  __attribute__((ext_vector_type(8)));
typedef float  v8f   __attribute__((ext_vector_type(8)));

__device__ __forceinline__ float sigmoidf_(float x) { return 1.0f / (1.0f + __expf(-x)); }

// ---------------------------------------------------------------------------
// fp32 -> bf16 conversion (weights) and zero-init helpers
// ---------------------------------------------------------------------------
__global__ void cvt_f32_bf16_kernel(const float* __restrict__ src, __bf16* __restrict__ dst, int n) {
  int i = blockIdx.x * blockDim.x + threadIdx.x;
  if (i < n) dst[i] = (__bf16)src[i];
}

__global__ void zero_bf16_kernel(__bf16* __restrict__ p, int n) {
  int i = blockIdx.x * blockDim.x + threadIdx.x;
  if (i < n) p[i] = (__bf16)0.f;
}

// ---------------------------------------------------------------------------
// Conv-attention pooling over P: single HBM pass, tile staged in LDS (bf16).
// scores = x . conv_w + b ; softmax over P ; conv_all = sum_p attn[p]*x[p]
// Output row order is (s*B + b) so GRU steps read contiguous 32-row blocks.
// ---------------------------------------------------------------------------
__global__ __launch_bounds__(256)
void conv_attn_kernel(const float* __restrict__ inputs,
                      const float* __restrict__ conv_w,
                      const float* __restrict__ conv_b,
                      __bf16* __restrict__ convA) {
  __shared__ __bf16 tile[P_ * I_];   // 32 KB
  __shared__ float  cw[I_];          // 2 KB
  __shared__ float  sc[P_];
  __shared__ float  aw[P_];
  const int tid = threadIdx.x;
  const int bs  = blockIdx.x;        // b*S + s
  const int b   = bs / S_, s = bs % S_;
  const size_t base = (size_t)bs * (P_ * I_);

  for (int i = tid; i < I_; i += 256) cw[i] = conv_w[i];
  for (int idx = tid * 4; idx < P_ * I_; idx += 256 * 4) {
    float4 v = *(const float4*)(inputs + base + idx);
    tile[idx + 0] = (__bf16)v.x; tile[idx + 1] = (__bf16)v.y;
    tile[idx + 2] = (__bf16)v.z; tile[idx + 3] = (__bf16)v.w;
  }
  __syncthreads();

  const int wave = tid >> 5, lane = tid & 31;
  for (int p = wave; p < P_; p += 8) {
    float sum = 0.f;
    for (int i = lane; i < I_; i += 32) sum += (float)tile[p * I_ + i] * cw[i];
    for (int o = 16; o > 0; o >>= 1) sum += __shfl_xor(sum, o);
    if (lane == 0) sc[p] = sum + conv_b[0];
  }
  __syncthreads();
  if (tid < 32) {                     // softmax over P=32 in one wave
    float v = sc[tid];
    float m = v;
    for (int o = 16; o > 0; o >>= 1) m = fmaxf(m, __shfl_xor(m, o));
    float e = __expf(v - m);
    float d = e;
    for (int o = 16; o > 0; o >>= 1) d += __shfl_xor(d, o);
    aw[tid] = e / d;
  }
  __syncthreads();
  for (int i = tid; i < I_; i += 256) {
    float acc = 0.f;
    #pragma unroll
    for (int p = 0; p < P_; ++p) acc += aw[p] * (float)tile[p * I_ + i];
    convA[(size_t)(s * B_ + b) * I_ + i] = (__bf16)acc;
  }
}

// ---------------------------------------------------------------------------
// Generic bf16 WMMA GEMM: C[M,N] = act(A[M,K] @ W[N,K]^T + bias)
// One wave owns a 16x16 C tile; K-loop of v_wmma_f32_16x16x32_bf16.
// mode 0: fp32 out (+bias)   mode 1: hardtanh clip -> bf16 out
// A layout per lane (16-bit A 16x32): row = lane&15,
//   K = 16*(lane>>4) .. +15 (contiguous, matching B).
// B layout (32x16, B[k][n] = W[n][k]): col = lane&15,
//   K = 16*(lane>>4) .. +15  -> one contiguous 32-byte load from W row.
// ---------------------------------------------------------------------------
__global__ __launch_bounds__(256)
void wmma_gemm_kernel(const __bf16* __restrict__ A,
                      const __bf16* __restrict__ W,
                      const float*  __restrict__ bias,
                      void* __restrict__ Cout,
                      int M, int N, int K, int mode) {
  const int lane   = threadIdx.x & 31;
  const int wid    = threadIdx.x >> 5;
  const int tilesN = N >> 4;
  const int tileId = blockIdx.x * (blockDim.x >> 5) + wid;
  if (tileId >= (M >> 4) * tilesN) return;
  const int mt = tileId / tilesN, nt = tileId % tilesN;

  const __bf16* aRow = A + (size_t)(mt * 16 + (lane & 15)) * K + ((lane >> 4) * 16);
  const __bf16* wRow = W + (size_t)(nt * 16 + (lane & 15)) * K + ((lane >> 4) * 16);

  v8f acc = {};
  for (int kb = 0; kb < K; kb += 32) {
    v8bf alo = *(const v8bf*)(aRow + kb);
    v8bf ahi = *(const v8bf*)(aRow + kb + 8);
    v16bf a;
    #pragma unroll
    for (int i = 0; i < 8; ++i) { a[i] = alo[i]; a[8 + i] = ahi[i]; }
    v16bf bm = *(const v16bf*)(wRow + kb);
    __builtin_prefetch(aRow + kb + 64, 0, 0);   // global_prefetch_b8: A slab two K-steps ahead
    acc = __builtin_amdgcn_wmma_f32_16x16x32_bf16(false, a, false, bm,
                                                  (short)0, acc, false, false);
  }
  const int   n  = nt * 16 + (lane & 15);
  const float bb = bias ? bias[n] : 0.f;
  #pragma unroll
  for (int v = 0; v < 8; ++v) {
    const int m = mt * 16 + v + ((lane >> 4) << 3);   // C layout: VGPR v, lane halves
    float c = acc[v] + bb;
    if (mode == 1) {
      c = fminf(1.f, fmaxf(-1.f, c));                 // Hardtanh
      ((__bf16*)Cout)[(size_t)m * N + n] = (__bf16)c;
    } else {
      ((float*)Cout)[(size_t)m * N + n] = c;
    }
  }
}

// ---------------------------------------------------------------------------
// One GRU time step (one layer): gh = h_prev @ W_hh^T + b_hh via WMMA,
// then fused PyTorch gate math. Block = 16 hidden columns, 6 waves =
// (2 M-tiles of batch 32) x (3 gates r/z/n); gate fusion via 6 KB LDS.
// gi already contains x_t @ W_ih^T + b_ih (hoisted big GEMM).
// ---------------------------------------------------------------------------
__global__ __launch_bounds__(192)
void gru_step_kernel(const float*  __restrict__ gi,      // [B,3H] for this step
                     const __bf16* __restrict__ Whh,     // [3H,H]
                     const float*  __restrict__ bhh,     // [3H]
                     const __bf16* __restrict__ hprev,   // [B,H]
                     __bf16* __restrict__ hbf_out,       // [B,H]
                     float*  __restrict__ yf32,          // [B,S,H] base or null
                     int t) {
  __shared__ float gh[3][B_][16];
  const int tid  = threadIdx.x;
  const int lane = tid & 31, wid = tid >> 5;   // 6 waves
  const int mt   = wid & 1, gate = wid >> 1;
  const int nb   = blockIdx.x * 16;

  const __bf16* aRow = hprev + (size_t)(mt * 16 + (lane & 15)) * H_ + ((lane >> 4) * 16);
  const __bf16* wRow = Whh + (size_t)(gate * H_ + nb + (lane & 15)) * H_ + ((lane >> 4) * 16);

  v8f acc = {};
  for (int kb = 0; kb < H_; kb += 32) {
    v8bf alo = *(const v8bf*)(aRow + kb);
    v8bf ahi = *(const v8bf*)(aRow + kb + 8);
    v16bf a;
    #pragma unroll
    for (int i = 0; i < 8; ++i) { a[i] = alo[i]; a[8 + i] = ahi[i]; }
    v16bf bm = *(const v16bf*)(wRow + kb);
    acc = __builtin_amdgcn_wmma_f32_16x16x32_bf16(false, a, false, bm,
                                                  (short)0, acc, false, false);
  }
  const int   nl = lane & 15;
  const float bb = bhh[gate * H_ + nb + nl];
  #pragma unroll
  for (int v = 0; v < 8; ++v) {
    const int m = mt * 16 + v + ((lane >> 4) << 3);
    gh[gate][m][nl] = acc[v] + bb;
  }
  __syncthreads();

  for (int idx = tid; idx < B_ * 16; idx += 192) {
    const int b = idx >> 4, nn = idx & 15;
    const int col = nb + nn;
    const float gr = gh[0][b][nn], gz = gh[1][b][nn], gn = gh[2][b][nn];
    const float ir  = gi[(size_t)b * 3 * H_ + col];
    const float iz  = gi[(size_t)b * 3 * H_ + H_ + col];
    const float inn = gi[(size_t)b * 3 * H_ + 2 * H_ + col];
    const float r = sigmoidf_(ir + gr);
    const float z = sigmoidf_(iz + gz);
    const float n = tanhf(inn + r * gn);
    const float hp = (float)hprev[(size_t)b * H_ + col];
    const float hn = (1.f - z) * n + z * hp;
    hbf_out[(size_t)b * H_ + col] = (__bf16)hn;
    if (yf32) yf32[((size_t)b * S_ + t) * H_ + col] = hn;
  }
}

// ---------------------------------------------------------------------------
// Attention over time: e = states . att_w, softmax over S, context = sum.
// ---------------------------------------------------------------------------
__global__ __launch_bounds__(256)
void att_time_kernel(const float* __restrict__ states,  // [B,S,H]
                     const float* __restrict__ att_w,   // [H]
                     float* __restrict__ context,       // [B,H]
                     float* __restrict__ alpha) {       // [B,S]
  __shared__ float e[S_];
  const int b = blockIdx.x;
  const int tid = threadIdx.x, lane = tid & 31, wid = tid >> 5;
  const float* st = states + (size_t)b * S_ * H_;
  for (int s = wid; s < S_; s += 8) {
    float sum = 0.f;
    for (int h = lane; h < H_; h += 32) sum += st[(size_t)s * H_ + h] * att_w[h];
    for (int o = 16; o > 0; o >>= 1) sum += __shfl_xor(sum, o);
    if (lane == 0) e[s] = sum;
  }
  __syncthreads();
  if (tid == 0) {
    float m = -1e30f;
    for (int s = 0; s < S_; ++s) m = fmaxf(m, e[s]);
    float d = 0.f;
    for (int s = 0; s < S_; ++s) { e[s] = __expf(e[s] - m); d += e[s]; }
    const float inv = 1.f / d;
    for (int s = 0; s < S_; ++s) e[s] *= inv;
  }
  __syncthreads();
  if (tid < S_) alpha[(size_t)b * S_ + tid] = e[tid];
  for (int h = tid; h < H_; h += 256) {
    float acc = 0.f;
    for (int s = 0; s < S_; ++s) acc += e[s] * st[(size_t)s * H_ + h];
    context[(size_t)b * H_ + h] = acc;
  }
}

// ---------------------------------------------------------------------------
// Final linear over [context, demoip] + softmax. Tiny: one wave, lane = batch.
// ---------------------------------------------------------------------------
__global__ void final_kernel(const float* __restrict__ context,  // [B,H]
                             const float* __restrict__ demoip,   // [B,3]
                             const float* __restrict__ linW,     // [O,H+3]
                             const float* __restrict__ linb,     // [O]
                             float* __restrict__ out) {          // [B,O]
  const int b = threadIdx.x;
  if (b >= B_) return;
  float logit[O_];
  float m = -1e30f;
  for (int o = 0; o < O_; ++o) {
    const float* w = linW + (size_t)o * (H_ + 3);
    float acc = linb[o];
    for (int h = 0; h < H_; ++h) acc += context[(size_t)b * H_ + h] * w[h];
    acc += demoip[b * 3 + 0] * w[H_ + 0] + demoip[b * 3 + 1] * w[H_ + 1] +
           demoip[b * 3 + 2] * w[H_ + 2];
    logit[o] = acc;
    m = fmaxf(m, acc);
  }
  float d = 0.f;
  for (int o = 0; o < O_; ++o) { logit[o] = __expf(logit[o] - m); d += logit[o]; }
  const float inv = 1.f / d;
  for (int o = 0; o < O_; ++o) out[(size_t)b * O_ + o] = logit[o] * inv;
}

// ---------------------------------------------------------------------------
extern "C" void kernel_launch(void* const* d_in, const int* in_sizes, int n_in,
                              void* d_out, int out_size, void* d_ws, size_t ws_size,
                              hipStream_t stream) {
  (void)in_sizes; (void)n_in; (void)out_size; (void)ws_size;

  const float* inputs  = (const float*)d_in[0];
  const float* demoip  = (const float*)d_in[1];
  /* d_in[2] = batch_size (unused) */
  const float* conv_w  = (const float*)d_in[3];
  const float* conv_b  = (const float*)d_in[4];
  const float* embed_W = (const float*)d_in[5];
  const float* Wih0    = (const float*)d_in[6];
  const float* Whh0    = (const float*)d_in[7];
  const float* bih0    = (const float*)d_in[8];
  const float* bhh0    = (const float*)d_in[9];
  const float* Wih1    = (const float*)d_in[10];
  const float* Whh1    = (const float*)d_in[11];
  const float* bih1    = (const float*)d_in[12];
  const float* bhh1    = (const float*)d_in[13];
  const float* att_w   = (const float*)d_in[14];
  const float* lin_W   = (const float*)d_in[15];
  const float* lin_b   = (const float*)d_in[16];

  // Output tuple, flat in return order: out[B,O], states[B,S,H], context[B,H], alpha[B,S]
  float* out     = (float*)d_out;
  float* states  = out + B_ * O_;
  float* context = states + (size_t)B_ * S_ * H_;
  float* alpha   = context + (size_t)B_ * H_;

  // Workspace carve-up (~45 MB), 256B aligned slices
  char* p = (char*)d_ws;
  auto take = [&](size_t bytes) { char* r = p; p += (bytes + 255) & ~(size_t)255; return r; };
  __bf16* wEmbB = (__bf16*)take((size_t)E_ * I_ * 2);
  __bf16* wIh0B = (__bf16*)take((size_t)3 * H_ * E_ * 2);
  __bf16* wHh0B = (__bf16*)take((size_t)3 * H_ * H_ * 2);
  __bf16* wIh1B = (__bf16*)take((size_t)3 * H_ * H_ * 2);
  __bf16* wHh1B = (__bf16*)take((size_t)3 * H_ * H_ * 2);
  __bf16* convA = (__bf16*)take((size_t)S_ * B_ * I_ * 2);   // [S*B, I] bf16
  __bf16* embB  = (__bf16*)take((size_t)S_ * B_ * E_ * 2);   // [S*B, E] bf16 (post-hardtanh)
  __bf16* y0    = (__bf16*)take((size_t)S_ * B_ * H_ * 2);   // layer-0 states, bf16
  float*  gi    = (float*) take((size_t)S_ * B_ * 3 * H_ * 4); // hoisted x@W_ih^T+b (reused)
  __bf16* hb0   = (__bf16*)take((size_t)B_ * H_ * 2);
  __bf16* hb1   = (__bf16*)take((size_t)B_ * H_ * 2);
  __bf16* zb    = (__bf16*)take((size_t)B_ * H_ * 2);

  auto cvt = [&](const float* s, __bf16* d, int n) {
    cvt_f32_bf16_kernel<<<(n + 255) / 256, 256, 0, stream>>>(s, d, n);
  };
  cvt(embed_W, wEmbB, E_ * I_);
  cvt(Wih0, wIh0B, 3 * H_ * E_);
  cvt(Whh0, wHh0B, 3 * H_ * H_);
  cvt(Wih1, wIh1B, 3 * H_ * H_);
  cvt(Whh1, wHh1B, 3 * H_ * H_);
  zero_bf16_kernel<<<(B_ * H_ + 255) / 256, 256, 0, stream>>>(zb, B_ * H_);

  // 1) Conv-attn pooling (single HBM pass over 256 MB input)
  conv_attn_kernel<<<B_ * S_, 256, 0, stream>>>(inputs, conv_w, conv_b, convA);

  const int M = S_ * B_;   // 4096
  // 2) emb = hardtanh(conv_all @ embed_W^T) -> bf16
  wmma_gemm_kernel<<<((M / 16) * (E_ / 16)) / 8, 256, 0, stream>>>(
      convA, wEmbB, nullptr, embB, M, E_, I_, 1);
  // 3) Layer-0 input gates, hoisted over all timesteps
  wmma_gemm_kernel<<<((M / 16) * (3 * H_ / 16)) / 8, 256, 0, stream>>>(
      embB, wIh0B, bih0, gi, M, 3 * H_, E_, 0);
  // 4) Layer-0 recurrence: h written bf16 into y0 (doubles as layer-1 GEMM input)
  for (int t = 0; t < S_; ++t) {
    const __bf16* hp = (t == 0) ? zb : (y0 + (size_t)(t - 1) * B_ * H_);
    gru_step_kernel<<<H_ / 16, 192, 0, stream>>>(
        gi + (size_t)t * B_ * 3 * H_, wHh0B, bhh0, hp, y0 + (size_t)t * B_ * H_, nullptr, t);
  }
  // 5) Layer-1 input gates (reuse gi buffer)
  wmma_gemm_kernel<<<((M / 16) * (3 * H_ / 16)) / 8, 256, 0, stream>>>(
      y0, wIh1B, bih1, gi, M, 3 * H_, H_, 0);
  // 6) Layer-1 recurrence: fp32 states straight into d_out, bf16 ping-pong for recurrence
  for (int t = 0; t < S_; ++t) {
    const __bf16* hp = (t == 0) ? zb : ((t & 1) ? hb1 : hb0);
    __bf16* hn = (t & 1) ? hb0 : hb1;
    gru_step_kernel<<<H_ / 16, 192, 0, stream>>>(
        gi + (size_t)t * B_ * 3 * H_, wHh1B, bhh1, hp, hn, states, t);
  }
  // 7) Time attention + final linear/softmax
  att_time_kernel<<<B_, 256, 0, stream>>>(states, att_w, context, alpha);
  final_kernel<<<1, 32, 0, stream>>>(context, demoip, lin_W, lin_b, out);
}
